// BiMamba_74466142978823
// MI455X (gfx1250) — matrix operand order repaired and checked by reading, not compiled, not run on hardware
//
#include <hip/hip_runtime.h>
#include <cstdint>

// ---------------------------------------------------------------------------
// BiMamba on gfx1250 (MI455X).
//  - All GEMMs run on v_wmma_f32_16x16x32_bf16 (f32 in memory, bf16 MACs,
//    f32 accumulate), LDS-staged with packed v_perm bf16 conversion and
//    b64 DS stores; global_prefetch for the next K tile.
//  - Selective scan: 1 thread per (b,d) channel, 16 states in registers.
// ---------------------------------------------------------------------------

typedef __attribute__((ext_vector_type(16))) __bf16 v16bf;
typedef __attribute__((ext_vector_type(8)))  __bf16 bf16x8;
typedef __attribute__((ext_vector_type(8)))  float  v8f;

#define L_SEQ    1024
#define DMODEL   1024
#define DINNER   2048
#define NBATCH   2
#define NXDBL    96     // DT_RANK + 2*D_STATE
#define DTRANK   64
#define NSTATE   16

// Pack two f32 into two bf16 (round-half-up) with one v_perm_b32.
__device__ __forceinline__ uint32_t pk2bf(float a, float b) {
  uint32_t ua = __builtin_bit_cast(uint32_t, a) + 0x8000u;
  uint32_t ub = __builtin_bit_cast(uint32_t, b) + 0x8000u;
  // result bytes: {ub[3],ub[2],ua[3],ua[2]} -> low half = bf16(a), high = bf16(b)
  return __builtin_amdgcn_perm(ub, ua, 0x07060302u);
}

union FragU { v16bf v; bf16x8 h[2]; };

// ---------------------------------------------------------------------------
// Tiled WMMA GEMM:  C[M,N] = A[M,K] (row-major f32) * B  (+ epilogue)
//   BMODE 0: B element (k,n) at B0[n*ldb + k]       (n-major)
//   BMODE 1: B element (k,n) at B0[k*ldb + n]       (k-major)
//   BMODE 2: B element (k,n) = 0.5*(B0[k*ldb+n] + B1[k*ldb + (N-1-n)])
//   CMODE 0: store C[m*ldc + n];  CMODE 1: store C[n*ldc + m]
//   EPI   1: v = softplus(v + bias[m])
//   MCHECK : bounds-check M (only needed when M % 128 != 0)
// Requires: N % 128 == 0, K % 32 == 0, lda % 4 == 0.
// Block tile 128x128, 8 waves (2x4), wave tile 64x32 = 4x2 WMMA tiles, TK=32.
// ---------------------------------------------------------------------------
template<int BMODE, int CMODE, int EPI, int MCHECK>
__global__ __launch_bounds__(256) void wmma_gemm(
    const float* __restrict__ A, const float* __restrict__ B0,
    const float* __restrict__ B1, const float* __restrict__ bias,
    float* __restrict__ C, int M, int N, int K,
    int lda, int ldb, int ldc,
    size_t strideA, size_t strideB, size_t strideC)
{
  constexpr int TM = 128, TN = 128, TK = 32;
  __shared__ __bf16 As[TM][TK + 8];   // [m][k]
  __shared__ __bf16 Bs[TN][TK + 8];   // [n][k]

  const int tid  = threadIdx.x;
  const int lane = tid & 31;
  const int wave = tid >> 5;
  const int wm   = wave >> 2;              // 0..1 (M)
  const int wn   = wave & 3;               // 0..3 (N)
  const int blockM = blockIdx.y * TM;
  const int blockN = blockIdx.x * TN;

  const float* Ab  = A  + strideA * blockIdx.z;
  const float* B0b = B0 + strideB * blockIdx.z;
  const float* B1b = (BMODE == 2) ? (B1 + strideB * blockIdx.z) : nullptr;
  float*       Cb  = C  + strideC * blockIdx.z;

  v8f acc[4][2] = {};

  for (int k0 = 0; k0 < K; k0 += TK) {
    // ---- prefetch next K tile (global_prefetch_b8) ----
    if (k0 + TK < K) {
      __builtin_prefetch(Ab + (size_t)(blockM + (tid >> 1)) * lda + (k0 + TK),
                         0, 1);
      if constexpr (BMODE == 0)
        __builtin_prefetch(B0b + (size_t)(blockN + (tid >> 1)) * ldb + (k0 + TK),
                           0, 1);
      else
        __builtin_prefetch(B0b + (size_t)(k0 + TK + (tid >> 5)) * ldb +
                               (blockN + (tid & 31) * 4), 0, 1);
    }

    // ---- stage A tile: float4 global loads, packed b64 DS stores ----
#pragma unroll
    for (int j = 0; j < (TM * (TK / 4)) / 256; ++j) {      // 4 iters
      int idx = j * 256 + tid;
      int r = idx >> 3, kc = idx & 7;                       // row, 4-wide k chunk
      int m = blockM + r;
      float4 v;
      if (MCHECK && m >= M) v = make_float4(0.f, 0.f, 0.f, 0.f);
      else v = *(const float4*)(Ab + (size_t)m * lda + (k0 + kc * 4));
      *reinterpret_cast<uint2*>(&As[r][kc * 4]) =
          make_uint2(pk2bf(v.x, v.y), pk2bf(v.z, v.w));
    }
    // ---- stage B tile as Bs[n][k] ----
    if constexpr (BMODE == 0) {
#pragma unroll
      for (int j = 0; j < (TN * (TK / 4)) / 256; ++j) {
        int idx = j * 256 + tid;
        int r = idx >> 3, kc = idx & 7;
        float4 v = *(const float4*)(B0b + (size_t)(blockN + r) * ldb +
                                    (k0 + kc * 4));
        *reinterpret_cast<uint2*>(&Bs[r][kc * 4]) =
            make_uint2(pk2bf(v.x, v.y), pk2bf(v.z, v.w));
      }
    } else {
#pragma unroll
      for (int j = 0; j < (TN * (TK / 4)) / 256; ++j) {
        int idx = j * 256 + tid;
        int n = idx & (TN - 1), kc = idx >> 7;              // n fastest: coalesced
        int ng = blockN + n;
        float e[4];
#pragma unroll
        for (int q = 0; q < 4; ++q) {
          int k = k0 + kc * 4 + q;
          float v = B0b[(size_t)k * ldb + ng];
          if constexpr (BMODE == 2)
            v = 0.5f * (v + B1b[(size_t)k * ldb + (N - 1 - ng)]);
          e[q] = v;
        }
        *reinterpret_cast<uint2*>(&Bs[n][kc * 4]) =
            make_uint2(pk2bf(e[0], e[1]), pk2bf(e[2], e[3]));
      }
    }
    __syncthreads();

    // ---- fragment loads per ISA 16-bit A/B lane layouts ----
    // A 16x32: lane m = lane&15; K chunk {kb..kb+7, 16+kb..16+kb+7}, kb=(lane>>4)*8
    v16bf afrag[4];
    const int kb = (lane >> 4) * 8;
#pragma unroll
    for (int mi = 0; mi < 4; ++mi) {
      FragU f;
      const __bf16* p = &As[wm * 64 + mi * 16 + (lane & 15)][0];
      f.h[0] = *(const bf16x8*)(p + kb);
      f.h[1] = *(const bf16x8*)(p + 16 + kb);
      afrag[mi] = f.v;
    }
    // B 32x16: lane n = lane&15; K chunk {kb..kb+7, 16+kb..16+kb+7}, kb=(lane>>4)*8
    v16bf bfrag[2];
#pragma unroll
    for (int ni = 0; ni < 2; ++ni) {
      FragU f;
      const __bf16* p = &Bs[wn * 32 + ni * 16 + (lane & 15)][0];
      f.h[0] = *(const bf16x8*)(p + kb);
      f.h[1] = *(const bf16x8*)(p + 16 + kb);
      bfrag[ni] = f.v;
    }
#pragma unroll
    for (int mi = 0; mi < 4; ++mi)
#pragma unroll
      for (int ni = 0; ni < 2; ++ni)
        acc[mi][ni] = __builtin_amdgcn_wmma_f32_16x16x32_bf16(
            false, afrag[mi], false, bfrag[ni], (short)0, acc[mi][ni],
            false, false);
    __syncthreads();
  }

  // ---- epilogue + store (C layout: lane n = lane&15, m = (lane>=16)*8 + r)
#pragma unroll
  for (int mi = 0; mi < 4; ++mi) {
#pragma unroll
    for (int ni = 0; ni < 2; ++ni) {
#pragma unroll
      for (int r = 0; r < 8; ++r) {
        int mg = blockM + wm * 64 + mi * 16 + ((lane >> 4) << 3) + r;
        int ng = blockN + wn * 32 + ni * 16 + (lane & 15);
        if (!MCHECK || mg < M) {
          float v = acc[mi][ni][r];
          if constexpr (EPI == 1) {
            v += bias[mg];
            v = (v > 20.f) ? v : log1pf(__expf(v));   // softplus
          }
          if constexpr (CMODE == 0)
            Cb[(size_t)mg * ldc + ng] = v;
          else
            Cb[(size_t)ng * ldc + mg] = v;
        }
      }
    }
  }
}

// ---------------------------------------------------------------------------
// Depthwise causal conv (width 4) + bias + SiLU.
// reverse!=0: operates on the time-reversed sequence (backward branch works
// entirely in reversed-time coordinates t, matching xz[..., ::-1]).
// ---------------------------------------------------------------------------
__global__ __launch_bounds__(256) void conv_silu_kernel(
    const float* __restrict__ xz, const float* __restrict__ cw,
    const float* __restrict__ cb, float* __restrict__ xc, int reverse)
{
  int idx  = blockIdx.x * 256 + threadIdx.x;
  int t    = idx % L_SEQ;
  int rest = idx / L_SEQ;
  int d    = rest % DINNER;
  int b    = rest / DINNER;
  const size_t xbase = (size_t)(b * 2 * DINNER + d) * L_SEQ;  // x half of xz
  float acc = cb[d];
#pragma unroll
  for (int j = 0; j < 4; ++j) {
    int tt = t - 3 + j;
    if (tt >= 0) {
      int lo = reverse ? (L_SEQ - 1 - tt) : tt;
      acc += cw[d * 4 + j] * xz[xbase + lo];
    }
  }
  xc[(size_t)(b * DINNER + d) * L_SEQ + t] = acc / (1.f + __expf(-acc));
}

// ---------------------------------------------------------------------------
// Selective scan: one thread per (b,d). 16-state recurrence in registers.
//   h[n] = exp(delta*A[n]) * h[n] + delta*x*B[n] ;  y = sum_n h[n]*C[n]
//   out  = (y + x*D) * silu(z)
// xdbl is [b][l][96] (transposed GEMM output) so B/C are contiguous 16-float
// reads per timestep, uniform across the wave.
// ---------------------------------------------------------------------------
__global__ __launch_bounds__(256) void scan_kernel(
    const float* __restrict__ delta, const float* __restrict__ xc,
    const float* __restrict__ xdbl,  const float* __restrict__ xz,
    const float* __restrict__ A_log, const float* __restrict__ Dp,
    float* __restrict__ y, int reverse)
{
  const int g = blockIdx.x * 256 + threadIdx.x;
  const int b = g / DINNER;
  const int d = g % DINNER;

  float An[NSTATE], h[NSTATE];
#pragma unroll
  for (int n = 0; n < NSTATE; ++n) {
    An[n] = -__expf(A_log[d * NSTATE + n]);
    h[n]  = 0.f;
  }
  const float  Dv    = Dp[d];
  const size_t rbase = (size_t)(b * DINNER + d) * L_SEQ;
  const size_t zbase = (size_t)(b * 2 * DINNER + DINNER + d) * L_SEQ;
  const size_t xdbase = (size_t)b * L_SEQ * NXDBL;

  for (int l0 = 0; l0 < L_SEQ; l0 += 4) {
    float4 d4 = *(const float4*)(delta + rbase + l0);
    float4 x4 = *(const float4*)(xc + rbase + l0);
    float4 z4;
    if (!reverse) {
      z4 = *(const float4*)(xz + zbase + l0);
    } else {
      float4 t = *(const float4*)(xz + zbase + (L_SEQ - 4 - l0));
      z4 = make_float4(t.w, t.z, t.y, t.x);
    }
    float dl[4] = {d4.x, d4.y, d4.z, d4.w};
    float xv[4] = {x4.x, x4.y, x4.z, x4.w};
    float zv[4] = {z4.x, z4.y, z4.z, z4.w};
    float o4[4];
#pragma unroll
    for (int i = 0; i < 4; ++i) {
      const int l = l0 + i;
      const float4* B4 =
          (const float4*)(xdbl + xdbase + (size_t)l * NXDBL + DTRANK);
      float Bn[NSTATE], Cn[NSTATE];
#pragma unroll
      for (int q = 0; q < 4; ++q) {
        float4 tb = B4[q];
        Bn[4*q+0]=tb.x; Bn[4*q+1]=tb.y; Bn[4*q+2]=tb.z; Bn[4*q+3]=tb.w;
        float4 tc = B4[q + 4];
        Cn[4*q+0]=tc.x; Cn[4*q+1]=tc.y; Cn[4*q+2]=tc.z; Cn[4*q+3]=tc.w;
      }
      const float dx = dl[i] * xv[i];
      float acc = 0.f;
#pragma unroll
      for (int n = 0; n < NSTATE; ++n) {
        float dA = __expf(dl[i] * An[n]);
        h[n] = dA * h[n] + dx * Bn[n];
        acc += h[n] * Cn[n];
      }
      float sz = zv[i] / (1.f + __expf(-zv[i]));
      o4[i] = (acc + xv[i] * Dv) * sz;
    }
    *(float4*)(y + rbase + l0) = make_float4(o4[0], o4[1], o4[2], o4[3]);
  }
}

// ---------------------------------------------------------------------------
// Launch: in_proj GEMM -> per-direction {conv, x_proj GEMM, dt GEMM(+softplus),
// scan} -> fused out_proj GEMM with (fwd + reversed-bwd)/2 combine.
// ---------------------------------------------------------------------------
extern "C" void kernel_launch(void* const* d_in, const int* in_sizes, int n_in,
                              void* d_out, int out_size, void* d_ws, size_t ws_size,
                              hipStream_t stream) {
  (void)in_sizes; (void)n_in; (void)out_size; (void)ws_size;
  const float* hs   = (const float*)d_in[0];
  const float* Win  = (const float*)d_in[1];
  const float* Wout = (const float*)d_in[16];

  const float* cw[2] = {(const float*)d_in[2],  (const float*)d_in[9]};
  const float* cb[2] = {(const float*)d_in[3],  (const float*)d_in[10]};
  const float* xw[2] = {(const float*)d_in[4],  (const float*)d_in[11]};
  const float* dw[2] = {(const float*)d_in[5],  (const float*)d_in[12]};
  const float* db[2] = {(const float*)d_in[6],  (const float*)d_in[13]};
  const float* al[2] = {(const float*)d_in[7],  (const float*)d_in[14]};
  const float* Dd[2] = {(const float*)d_in[8],  (const float*)d_in[15]};
  float* out = (float*)d_out;

  float* ws    = (float*)d_ws;
  float* xz    = ws;                                     // B*4096*L
  float* xconv = xz    + (size_t)NBATCH * 2 * DINNER * L_SEQ;   // B*2048*L
  float* xdbl  = xconv + (size_t)NBATCH * DINNER * L_SEQ;       // B*L*96
  float* delta = xdbl  + (size_t)NBATCH * L_SEQ * NXDBL;        // B*2048*L
  float* y_f   = delta + (size_t)NBATCH * DINNER * L_SEQ;       // B*2048*L
  float* y_b   = y_f   + (size_t)NBATCH * DINNER * L_SEQ;       // B*2048*L
  float* ybuf[2] = {y_f, y_b};

  // 1) xz[b,e,l] = sum_d hs[b,l,d] * Win[e,d]
  wmma_gemm<0, 0, 0, 0><<<dim3(L_SEQ / 128, (2 * DINNER) / 128, NBATCH), 256, 0, stream>>>(
      Win, hs, nullptr, nullptr, xz,
      2 * DINNER, L_SEQ, DMODEL, DMODEL, DMODEL, L_SEQ,
      0, (size_t)L_SEQ * DMODEL, (size_t)2 * DINNER * L_SEQ);

  for (int dir = 0; dir < 2; ++dir) {
    // 2) depthwise causal conv + SiLU (reversed time for backward branch)
    conv_silu_kernel<<<(NBATCH * DINNER * L_SEQ) / 256, 256, 0, stream>>>(
        xz, cw[dir], cb[dir], xconv, dir);

    // 3) x_dbl[b,l,e96] = sum_d x_proj_w[e96,d] * xconv[b,d,l]  (C transposed)
    wmma_gemm<1, 1, 0, 1><<<dim3(L_SEQ / 128, 1, NBATCH), 256, 0, stream>>>(
        xw[dir], xconv, nullptr, nullptr, xdbl,
        NXDBL, L_SEQ, DINNER, DINNER, L_SEQ, NXDBL,
        0, (size_t)DINNER * L_SEQ, (size_t)L_SEQ * NXDBL);

    // 4) delta[b,d,l] = softplus(sum_r dt_w[d,r]*dtr[b,l,r] + dt_b[d])
    wmma_gemm<0, 0, 1, 0><<<dim3(L_SEQ / 128, DINNER / 128, NBATCH), 256, 0, stream>>>(
        dw[dir], xdbl, nullptr, db[dir], delta,
        DINNER, L_SEQ, DTRANK, DTRANK, NXDBL, L_SEQ,
        0, (size_t)L_SEQ * NXDBL, (size_t)DINNER * L_SEQ);

    // 5) selective scan + x*D + silu(z) gate
    scan_kernel<<<(NBATCH * DINNER) / 256, 256, 0, stream>>>(
        delta, xconv, xdbl, xz, al[dir], Dd[dir], ybuf[dir], dir);
  }

  // 6) out[b,l,o] = sum_d Wout[o,d] * 0.5*(y_f[b,d,l] + y_b[b,d,L-1-l])
  wmma_gemm<2, 1, 0, 0><<<dim3(L_SEQ / 128, DMODEL / 128, NBATCH), 256, 0, stream>>>(
      Wout, y_f, y_b, nullptr, out,
      DMODEL, L_SEQ, DINNER, DINNER, L_SEQ, DMODEL,
      0, (size_t)DINNER * L_SEQ, (size_t)L_SEQ * DMODEL);
}
